// GCN_simple_76398878261872
// MI455X (gfx1250) — matrix operand order repaired and checked by reading, not compiled, not run on hardware
//
#include <hip/hip_runtime.h>
#include <hip/hip_bf16.h>

// ---------------------------------------------------------------------------
// GCN forward for MI455X (gfx1250, wave32, WMMA).
//   h1  = relu(adj @ (v @ W1)), h2 = relu(adj @ (h1 @ W2))
//   out = (sum_n h2) @ W_out + b_out
// HBM-bound on adj (fp32, 256 MB). Strategy:
//  * adj -> bf16 in registers, all GEMMs via v_wmma_f32_16x16x32_bf16
//  * activations feeding adj-GEMMs stored transposed -> B stages to LDS as
//    pure b128 copies using GLOBAL_LOAD_ASYNC_TO_LDS (double buffered,
//    one barrier per K-step, s_wait_asynccnt gate)
//  * A fragments load straight from global, software-pipelined one K-tile
//    ahead so the cvt/wmma never stalls on loadcnt at the barrier
//  * 32-graph chunks: layer1 and layer2 over the same adj chunk (128 MB)
//    run back-to-back so the second adj pass hits the 192 MB L2
// ---------------------------------------------------------------------------

typedef __bf16 bf16_t;
typedef __attribute__((ext_vector_type(16))) __bf16 bf16x16;
typedef __attribute__((ext_vector_type(8)))  float  f32x8;
typedef __attribute__((ext_vector_type(4)))  int    v4i;

typedef __attribute__((address_space(1))) v4i gv4i;   // global int4
typedef __attribute__((address_space(3))) v4i sv4i;   // LDS int4

union Frag16 { bf16x16 v; uint4 q[2]; };
union Pack8  { uint4 q; bf16_t h[8]; };

#define TILE_K     32
#define BLOCK_ROWS 64
#define NT         8                // 8 col tiles of 16 -> N = 128
#define BSTRIDE    (128 * TILE_K)   // LDS elements per B buffer (8 KB)
#define GCHUNK     32               // graphs per L2-resident adj chunk

#ifndef __has_builtin
#define __has_builtin(x) 0
#endif

#if __has_builtin(__builtin_amdgcn_global_load_async_to_lds_b128)
#define USE_ASYNC 1
#else
#define USE_ASYNC 0
#endif

#if __has_builtin(__builtin_amdgcn_s_wait_asynccnt)
#define WAIT_ASYNC0() __builtin_amdgcn_s_wait_asynccnt(0)
#else
#define WAIT_ASYNC0() asm volatile("s_wait_asynccnt 0x0" ::: "memory")
#endif

// Copy 64 contiguous bytes global -> LDS (per thread), async when available.
__device__ __forceinline__ void stage64(const bf16_t* __restrict__ src,
                                        bf16_t* __restrict__ dst)
{
#if USE_ASYNC
#pragma unroll
    for (int i = 0; i < 4; ++i)
        __builtin_amdgcn_global_load_async_to_lds_b128(
            (gv4i*)(src + i * 8), (sv4i*)(dst + i * 8), 0, 0);
#else
#pragma unroll
    for (int i = 0; i < 4; ++i)
        *(uint4*)(dst + i * 8) = *(const uint4*)(src + i * 8);
#endif
}

// C = act( A (M x K) @ B^T-stored (128 x K col-major) ), N = 128.
//   A_F32: A is fp32 (convert to bf16 in regs), else bf16.
//   OUT_T: write C transposed per-graph ([feature][node], node blocks of 1024),
//          else row-major (M x 128) bf16.
// A row stride == B col stride == K for every GEMM in this chain.
template<bool A_F32, bool RELU, bool OUT_T>
__global__ __launch_bounds__(128)
void gemm_wmma(const void* __restrict__ Ag_,
               const bf16_t* __restrict__ Bt,
               bf16_t* __restrict__ Cg,
               int K, long long sA, long long sB, long long sC)
{
    __shared__ __align__(16) bf16_t Bld[2 * BSTRIDE];   // 16 KB double buffer

    const int t    = threadIdx.x;
    const int lane = t & 31;
    const int wave = t >> 5;          // 4 waves x 16 rows = 64 rows per block
    const int l15  = lane & 15;
    const int half = lane >> 4;       // K sub-range selector per ISA A/B layout
    const int b    = blockIdx.y;
    const int rowBlock = blockIdx.x * BLOCK_ROWS;
    const long long aRow = rowBlock + wave * 16 + l15;

    const bf16_t* Bb = Bt + (long long)b * sB;

    f32x8 acc[NT];
    const f32x8 zero = {0.f,0.f,0.f,0.f,0.f,0.f,0.f,0.f};
#pragma unroll
    for (int n = 0; n < NT; ++n) acc[n] = zero;

    const int kTiles = K / TILE_K;

    // Prologue: stage B tile 0 (thread t owns column t: 64 B) and issue the
    // raw A loads for tile 0.
    stage64(Bb + (long long)t * K, &Bld[t * TILE_K]);

    float4 r0, r1, r2, r3;            // raw A (fp32 path)
    uint4  rq0, rq1;                  // raw A (bf16 path)
    if constexpr (A_F32) {
        const float* Ar = (const float*)Ag_ + (long long)b * sA + aRow * K;
        r0 = *(const float4*)(Ar + half * 8);
        r1 = *(const float4*)(Ar + half * 8 + 4);
        r2 = *(const float4*)(Ar + 16 + half * 8);
        r3 = *(const float4*)(Ar + 16 + half * 8 + 4);
    } else {
        const bf16_t* Ar = (const bf16_t*)Ag_ + (long long)b * sA + aRow * K;
        rq0 = *(const uint4*)(Ar + half * 8);
        rq1 = *(const uint4*)(Ar + 16 + half * 8);
    }

    for (int kt = 0; kt < kTiles; ++kt) {
        const int k0 = kt * TILE_K;

#if USE_ASYNC
        WAIT_ASYNC0();              // this wave's pending B stage (tile kt) done
#endif
        __syncthreads();            // all waves' stages visible / readers done

        // Stage next B tile into the other buffer (single-barrier double
        // buffering: previous readers of that buffer are past the barrier).
        if (kt + 1 < kTiles)
            stage64(Bb + (long long)t * K + (k0 + TILE_K),
                    &Bld[((kt + 1) & 1) * BSTRIDE + t * TILE_K]);

        // Issue raw A loads for tile kt+1 now; they complete during the 8
        // WMMAs below and are consumed next iteration (no loadcnt stall).
        float4 n0 = r0, n1 = r1, n2 = r2, n3 = r3;
        uint4  nq0 = rq0, nq1 = rq1;
        if constexpr (A_F32) {
            if (kt + 1 < kTiles) {
                const float* An = (const float*)Ag_ + (long long)b * sA
                                + aRow * K + (k0 + TILE_K);
                n0 = *(const float4*)(An + half * 8);
                n1 = *(const float4*)(An + half * 8 + 4);
                n2 = *(const float4*)(An + 16 + half * 8);
                n3 = *(const float4*)(An + 16 + half * 8 + 4);
                __builtin_prefetch(An + TILE_K, 0, 3);
            }
        } else {
            if (kt + 1 < kTiles) {
                const bf16_t* An = (const bf16_t*)Ag_ + (long long)b * sA
                                 + aRow * K + (k0 + TILE_K);
                nq0 = *(const uint4*)(An + half * 8);
                nq1 = *(const uint4*)(An + 16 + half * 8);
                __builtin_prefetch(An + TILE_K, 0, 3);
            }
        }

        // ---- build A fragment (16x32 bf16, ISA layout) from raw tile kt ----
        Frag16 a;
        if constexpr (A_F32) {
            a.v[0]=(bf16_t)r0.x; a.v[1]=(bf16_t)r0.y; a.v[2]=(bf16_t)r0.z; a.v[3]=(bf16_t)r0.w;
            a.v[4]=(bf16_t)r1.x; a.v[5]=(bf16_t)r1.y; a.v[6]=(bf16_t)r1.z; a.v[7]=(bf16_t)r1.w;
            a.v[8]=(bf16_t)r2.x; a.v[9]=(bf16_t)r2.y; a.v[10]=(bf16_t)r2.z; a.v[11]=(bf16_t)r2.w;
            a.v[12]=(bf16_t)r3.x; a.v[13]=(bf16_t)r3.y; a.v[14]=(bf16_t)r3.z; a.v[15]=(bf16_t)r3.w;
        } else {
            a.q[0] = rq0; a.q[1] = rq1;
        }

        // ---- 8 WMMAs: B fragments from LDS [col][k] (lane = column) ----
        const bf16_t* Bbuf = &Bld[(kt & 1) * BSTRIDE];
#pragma unroll
        for (int n = 0; n < NT; ++n) {
            Frag16 bb;
            const bf16_t* p = Bbuf + (n * 16 + l15) * TILE_K + half * 8;
            bb.q[0] = *(const uint4*)(p);
            bb.q[1] = *(const uint4*)(p + 16);
            acc[n] = __builtin_amdgcn_wmma_f32_16x16x32_bf16(
                         false, a.v, false, bb.v, (short)0, acc[n], false, false);
        }

        r0 = n0; r1 = n1; r2 = n2; r3 = n3;
        rq0 = nq0; rq1 = nq1;
    }

    // ---- epilogue: C/D layout -> VGPR r is row (r | 8+r) per lane half ----
    const long long rowG = (long long)rowBlock + wave * 16 + half * 8;
    if constexpr (OUT_T) {
        // per-graph col-major: Cg[(graph*128 + col)*1024 + node], 8 nodes packed
        const long long graph = rowG >> 10;
        const long long node  = rowG & 1023;
#pragma unroll
        for (int n = 0; n < NT; ++n) {
            const int col = n * 16 + l15;
            Pack8 pk;
#pragma unroll
            for (int r = 0; r < 8; ++r) {
                float vv = acc[n][r];
                if constexpr (RELU) vv = vv > 0.f ? vv : 0.f;
                pk.h[r] = (bf16_t)vv;
            }
            *(uint4*)(Cg + ((graph * 128 + col) << 10) + node) = pk.q;
        }
    } else {
        bf16_t* Cb = Cg + (long long)b * sC;
#pragma unroll
        for (int n = 0; n < NT; ++n) {
            const int col = n * 16 + l15;
#pragma unroll
            for (int r = 0; r < 8; ++r) {
                float vv = acc[n][r];
                if constexpr (RELU) vv = vv > 0.f ? vv : 0.f;
                Cb[(rowG + r) * 128 + col] = (bf16_t)vv;
            }
        }
    }
}

// transpose-convert fp32 (K x 128) -> bf16 (128 x K) for the weights
__global__ void cvtT_f32_bf16(const float* __restrict__ src,
                              bf16_t* __restrict__ dst, int K)
{
    int i = blockIdx.x * blockDim.x + threadIdx.x;
    if (i < K * 128) {
        int r = i >> 7, c = i & 127;
        dst[c * K + r] = (bf16_t)src[i];
    }
}

// sum-pool over nodes: (B, 1024, 128) bf16 row-major -> (B, 128) fp32
// 256 threads/graph, b128 loads, LDS reduction over 16 node groups.
__global__ __launch_bounds__(256)
void pool_kernel(const bf16_t* __restrict__ h2, float* __restrict__ pool)
{
    __shared__ float red[16][128];
    const int b  = blockIdx.x;
    const int t  = threadIdx.x;
    const int fg = t & 15;        // feature group: 8 features (16 B)
    const int ng = t >> 4;        // node group 0..15
    const bf16_t* base = h2 + (long long)b * 1024 * 128 + fg * 8;

    float s[8] = {0.f,0.f,0.f,0.f,0.f,0.f,0.f,0.f};
    for (int n = ng; n < 1024; n += 16) {
        const uint4 q = *(const uint4*)(base + (long long)n * 128);
        const bf16_t* e = (const bf16_t*)&q;
#pragma unroll
        for (int i = 0; i < 8; ++i) s[i] += (float)e[i];
    }
#pragma unroll
    for (int i = 0; i < 8; ++i) red[ng][fg * 8 + i] = s[i];
    __syncthreads();
    if (t < 128) {
        float r = 0.f;
#pragma unroll
        for (int g = 0; g < 16; ++g) r += red[g][t];
        pool[b * 128 + t] = r;
    }
}

// out = pool @ W_out + b_out : (64,128) @ (128,10) fp32
__global__ void head_kernel(const float* __restrict__ pool,
                            const float* __restrict__ Wout,
                            const float* __restrict__ bout,
                            float* __restrict__ out)
{
    int t = blockIdx.x * blockDim.x + threadIdx.x;
    if (t >= 64 * 10) return;
    int row = t / 10, col = t % 10;
    float s = bout[col];
#pragma unroll 8
    for (int k = 0; k < 128; ++k) s += pool[row * 128 + k] * Wout[k * 10 + col];
    out[t] = s;
}

extern "C" void kernel_launch(void* const* d_in, const int* in_sizes, int n_in,
                              void* d_out, int out_size, void* d_ws, size_t ws_size,
                              hipStream_t stream)
{
    (void)in_sizes; (void)n_in; (void)out_size; (void)ws_size;

    const float* v   = (const float*)d_in[0];  // (64,1024,64)
    const float* adj = (const float*)d_in[1];  // (64,1024,1024)
    const float* W1  = (const float*)d_in[2];  // (64,128)
    const float* W2  = (const float*)d_in[3];  // (128,128)
    const float* Wo  = (const float*)d_in[4];  // (128,10)
    const float* bo  = (const float*)d_in[5];  // (10,)
    float* out = (float*)d_out;                // (64,10)

    // Workspace: W1t | W2t | Xt (X1t, then HW2t) | H (h1, then h2) | pool
    char* ws = (char*)d_ws;
    const size_t ACT = (size_t)65536 * 128 * 2;        // 16 MB bf16 activation
    bf16_t* W1t  = (bf16_t*)(ws);                       // 128 x 64  (16 KB)
    bf16_t* W2t  = (bf16_t*)(ws + 16384);               // 128 x 128 (32 KB)
    bf16_t* Xt   = (bf16_t*)(ws + 65536);               // per-graph 128 x 1024
    bf16_t* H    = (bf16_t*)(ws + 65536 + ACT);         // row-major 65536 x 128
    float*  pool = (float*)(ws + 65536 + 2 * ACT);      // 32 KB

    cvtT_f32_bf16<<<(8192  + 255) / 256, 256, 0, stream>>>(W1, W1t, 64);
    cvtT_f32_bf16<<<(16384 + 255) / 256, 256, 0, stream>>>(W2, W2t, 128);

    // X1t = (v @ W1)^T per graph   (M=65536, K=64, A fp32 -> transposed out)
    gemm_wmma<true, false, true><<<dim3(1024, 1), 128, 0, stream>>>(
        v, W1t, Xt, 64, 0LL, 0LL, 0LL);

    // Process graphs in chunks whose adj slice (GCHUNK*4MB = 128 MB) fits the
    // 192 MB L2: the second adj pass of each chunk hits L2, halving HBM reads.
    for (int g0 = 0; g0 < 64; g0 += GCHUNK) {
        const float*  adjC = adj + (long long)g0 * 1024 * 1024;
        bf16_t*       XtC  = Xt + (long long)g0 * 128 * 1024;
        bf16_t*       HC   = H  + (long long)g0 * 1024 * 128;

        // h1 = relu(adj @ X1)      (batched, M=1024, K=1024, row-major out)
        gemm_wmma<true, true, false><<<dim3(16, GCHUNK), 128, 0, stream>>>(
            adjC, XtC, HC, 1024, 1024LL * 1024, 128LL * 1024, 1024LL * 128);

        // HW2t = (h1 @ W2)^T       (M=GCHUNK*1024, K=128, A bf16 -> reuse Xt)
        gemm_wmma<false, false, true><<<dim3(GCHUNK * 1024 / 64, 1), 128, 0, stream>>>(
            HC, W2t, XtC, 128, 0LL, 0LL, 0LL);

        // h2 = relu(adj @ HW2)     (adj chunk now L2-resident; reuse H)
        gemm_wmma<true, true, false><<<dim3(16, GCHUNK), 128, 0, stream>>>(
            adjC, XtC, HC, 1024, 1024LL * 1024, 128LL * 1024, 1024LL * 128);
    }

    pool_kernel<<<64, 256, 0, stream>>>(H, pool);
    head_kernel<<<3, 256, 0, stream>>>(pool, Wo, bo, out);
}
